// Correlation_88175678587300
// MI455X (gfx1250) — matrix operand order repaired and checked by reading, not compiled, not run on hardware
//
#include <hip/hip_runtime.h>
#include <hip/hip_bf16.h>

typedef _Float16 f16;
typedef _Float16 v8h  __attribute__((ext_vector_type(8)));
typedef _Float16 v16h __attribute__((ext_vector_type(16)));
typedef float    v4f  __attribute__((ext_vector_type(4)));
typedef float    v8f  __attribute__((ext_vector_type(8)));

// Explicit LDS (address_space 3) types: keeps clang emitting ds_* ops (not flat_*)
typedef __attribute__((address_space(3))) char  lds_char;
typedef __attribute__((address_space(3))) f16   lds_f16;
typedef __attribute__((address_space(3))) float lds_float;
typedef __attribute__((address_space(3))) v8h   lds_v8h;
typedef __attribute__((address_space(3))) v4f   lds_v4f;

#define CC   256
#define HH   64
#define WW   96
#define BB   8
#define DD   21          // displacement steps per axis
#define HW   (HH*WW)     // 6144 floats: channel stride
#define ROWH 264         // Qt row stride in halfs (256 + 8 pad, 16B aligned)
#define ROWF 260         // Kraw row stride in floats (256 + 4 pad, 16B aligned)

// Dynamic LDS layout:
//   Qt    : f16 [96][ROWH]            input1 row, transposed, fp16        = 50688 B
//   Kraw0 : f32 [96][ROWF]            input2 row buf 0 (async target)     = 99840 B
//   Kraw1 : f32 [96][ROWF]            input2 row buf 1 (async target)     = 99840 B
//   Sb    : f32 [2][3][3][16][16]     banded S tiles                      = 18432 B
#define QT_BYTES   (96*ROWH*2)
#define KRAW_BYTES (96*ROWF*4)
#define SB_ELEMS   (2*3*3*256)
#define LDS_BYTES  (QT_BYTES + 2*KRAW_BYTES + SB_ELEMS*4)   // 268800 < 320K

__device__ __forceinline__ unsigned lds_addr(const volatile void __attribute__((address_space(3)))* p) {
    return (unsigned)(unsigned long long)p;   // 32-bit LDS-relative address
}

// Async transpose-scatter copy of one 256x96 f32 row: LDS[x*ROWF + c] = g[c*HW + x].
// Per-lane LDS destination VGPR => the async unit performs the transpose; wave
// issues and moves on (ASYNCcnt), global side stays fully coalesced (32 lanes =
// 128B contiguous per instruction).
__device__ __forceinline__ void async_copy_row_T(const float* __restrict__ gbase,
                                                 unsigned lds_base, int wave, int lane) {
    for (int c = wave; c < CC; c += 8) {
        unsigned go = (unsigned)(c * HW) * 4u;
        #pragma unroll
        for (int k = 0; k < 3; ++k) {
            int x = lane + 32 * k;
            unsigned ldsa = lds_base + (unsigned)(x * ROWF + c) * 4u;
            unsigned voff = go + (unsigned)(x * 4);
            asm volatile("global_load_async_to_lds_b32 %0, %1, %2"
                         :: "v"(ldsa), "v"(voff), "s"(gbase) : "memory");
        }
    }
}

__device__ __forceinline__ void wait_async_all() {
    asm volatile("s_wait_asynccnt 0x0" ::: "memory");
}

// 16 consecutive f32 from LDS -> v16h (B fragment half-row). 4x ds_load_b128 +
// v_cvt_pk_f16_f32 chain; the converts co-execute with the XDL WMMA pipe.
__device__ __forceinline__ v16h load_cvt16(const lds_float* p) {
    v4f f0 = ((const lds_v4f*)p)[0];
    v4f f1 = ((const lds_v4f*)p)[1];
    v4f f2 = ((const lds_v4f*)p)[2];
    v4f f3 = ((const lds_v4f*)p)[3];
    v16h r;
    #pragma unroll
    for (int q = 0; q < 4; ++q) {
        r[q]      = (f16)f0[q];
        r[q + 4]  = (f16)f1[q];
        r[q + 8]  = (f16)f2[q];
        r[q + 12] = (f16)f3[q];
    }
    return r;
}

__global__ __launch_bounds__(256) void
corr_wmma_async_kernel(const float* __restrict__ in1,
                       const float* __restrict__ in2,
                       float* __restrict__ out)
{
    extern __shared__ char smem[];
    lds_char*  sm3   = (lds_char*)smem;               // AS0 -> AS3
    lds_f16*   Qt    = (lds_f16*)sm3;
    lds_float* Kraw0 = (lds_float*)(sm3 + QT_BYTES);
    lds_float* Kraw1 = (lds_float*)(sm3 + QT_BYTES + KRAW_BYTES);
    lds_float* Sb    = (lds_float*)(sm3 + QT_BYTES + 2 * KRAW_BYTES);

    const int tid  = threadIdx.x;
    const int lane = tid & 31;
    const int wave = tid >> 5;
    const int y    = blockIdx.x % HH;
    const int b    = blockIdx.x / HH;

    const float* i1     = in1 + (size_t)b * CC * HW + (size_t)y * WW;
    const float* i2base = in2 + (size_t)b * CC * HW;

    const unsigned koff0 = lds_addr(Kraw0);
    const unsigned koff1 = lds_addr(Kraw1);

    // ---- Prologue: kick off async fetch of the first K row, then build Qt ----
    {
        int ys0 = y - 20;
        if (ys0 >= 0 && ys0 < HH)
            async_copy_row_T(i2base + (size_t)ys0 * WW, koff0, wave, lane);
    }
    for (int c = wave; c < CC; c += 8) {
        const float* src = i1 + (size_t)c * HW;
        #pragma unroll
        for (int k = 0; k < 3; ++k) {
            int x = lane + 32 * k;
            Qt[x * ROWH + c] = (f16)src[x];     // coalesced read, fp16 transpose
        }
    }
    wait_async_all();
    __syncthreads();

    const int hi = lane >> 4;       // half-wave select (ISA fragment layouts)
    const int ml = lane & 15;       // row (A) / col (B) within tile

    int cur = 0;
    for (int di = 0; di < DD; ++di) {
        const int  ys  = y + di * 2 - 20;
        const bool inb = (ys >= 0) && (ys < HH);
        const int  nxt = cur ^ 1;

        // Kick off async fetch of the NEXT row into the other buffer; it
        // overlaps all the WMMA work below (tracked on ASYNCcnt).
        {
            int ysn = ys + 2;
            if (di + 1 < DD && ysn >= 0 && ysn < HH)
                async_copy_row_T(i2base + (size_t)ysn * WW, cur ? koff0 : koff1, wave, lane);
        }

        if (inb) {
            const lds_float* Kr = cur ? Kraw1 : Kraw0;   // AS3 select -> stays ds_*
            // 14 banded tile-GEMMs: t = e*7 + r, r -> (I, J), |I-J| <= 1
            for (int t = wave; t < 14; t += 8) {
                const int e = t / 7;
                const int r = t % 7;
                const int I = (r + 1) / 3;      // 0,0,1,1,1,2,2
                const int J = r - 2 * I;        // 0,1,0,1,2,1,2

                const int xa = 2 * (16 * I + ml) + e;
                const int xb = 2 * (16 * J + ml) + e;
                const lds_f16*   arow = Qt + xa * ROWH;
                const lds_float* brow = Kr + xb * ROWF;

                v8f acc = {0.f, 0.f, 0.f, 0.f, 0.f, 0.f, 0.f, 0.f};
                #pragma unroll
                for (int kk = 0; kk < 8; ++kk) {        // K = 256 = 8 x 32
                    // A 16x32 f16: lane<16: K[0..15]; lane>=16: K[16..31]
                    v8h a0 = *(const lds_v8h*)(arow + kk * 32 + hi * 16);
                    v8h a1 = *(const lds_v8h*)(arow + kk * 32 + hi * 16 + 8);
                    v16h A;
                    #pragma unroll
                    for (int q = 0; q < 8; ++q) { A[q] = a0[q]; A[q + 8] = a1[q]; }
                    // B 32x16: lane<16: K[0..15]; lane>=16: K[16..31] (cvt from f32)
                    v16h Bf = load_cvt16(brow + kk * 32 + hi * 16);
                    acc = __builtin_amdgcn_wmma_f32_16x16x32_f16(
                        false, A, false, Bf, (short)0, acc, false, false);
                }

                lds_float* sb = Sb + (((e * 3 + I) * 3 + (J - I + 1)) * 256);
                #pragma unroll
                for (int r8 = 0; r8 < 8; ++r8)
                    sb[(r8 + 8 * hi) * 16 + ml] = acc[r8];  // C/D: M = r8+8*hi, N = ml
            }
        } else {
            for (int i = tid; i < SB_ELEMS; i += 256) Sb[i] = 0.f;
        }

        __syncthreads();            // Sb ready

        // ---- Output: 21 dx rows x 96 x, coalesced non-temporal float4 stores
        // (keep the L2-resident inputs from being evicted by write data) ----
        float* ob = out + (((size_t)b * 441 + (size_t)di * 21) * HH + y) * WW;
        for (int g = tid; g < 21 * 24; g += 256) {
            const int j  = g / 24;
            const int x0 = (g % 24) * 4;
            v4f v;
            #pragma unroll
            for (int k = 0; k < 4; ++k) {
                const int x  = x0 + k;
                const int e  = x & 1;
                const int p  = x >> 1;
                const int pp = p + j - 10;
                float val = 0.f;
                if (pp >= 0 && pp < 48) {
                    const int I  = p >> 4;
                    const int Jt = pp >> 4;
                    val = Sb[(((e * 3 + I) * 3 + (Jt - I + 1)) * 16 + (p & 15)) * 16 + (pp & 15)]
                          * (1.0f / 256.0f);
                }
                v[k] = val;
            }
            __builtin_nontemporal_store(v, (v4f*)(ob + (size_t)j * HW + x0));
        }

        wait_async_all();           // this wave's prefetch landed in LDS
        __syncthreads();            // publish Kraw[nxt] to all waves; Sb free
        cur = nxt;
    }
}

extern "C" void kernel_launch(void* const* d_in, const int* in_sizes, int n_in,
                              void* d_out, int out_size, void* d_ws, size_t ws_size,
                              hipStream_t stream) {
    (void)in_sizes; (void)n_in; (void)d_ws; (void)ws_size; (void)out_size;
    const float* in1 = (const float*)d_in[0];
    const float* in2 = (const float*)d_in[1];
    float* out = (float*)d_out;
    dim3 grid(BB * HH);          // one workgroup per (b, y)
    dim3 block(256);             // 8 wave32
    corr_wmma_async_kernel<<<grid, block, LDS_BYTES, stream>>>(in1, in2, out);
}
